// LocallyBandedSparseAttention_66348654788799
// MI455X (gfx1250) — matrix operand order repaired and checked by reading, not compiled, not run on hardware
//
#include <hip/hip_runtime.h>
#include <hip/hip_bf16.h>

// ---------------------------------------------------------------------------
// MI455X (gfx1250) locally-banded attention, bf16 WMMA pipeline.
// All matmuls run through v_wmma_f32_16x16x32_bf16 (f32 accumulate).
// GEMM tiles are double-buffered in LDS and staged with the CDNA5 async
// DMA path (global_load_async_to_lds_b128, ASYNCcnt) so WMMA compute
// overlaps the global->LDS copy of the next K-tile.
// ---------------------------------------------------------------------------

typedef __attribute__((ext_vector_type(16))) __bf16       v16bf;
typedef __attribute__((ext_vector_type(8)))  float        v8f;
typedef __attribute__((ext_vector_type(4)))  unsigned int v4u;

#define S_LEN 2048
#define DM    512
#define WIN   64
#define ATT_SCALE 0.125f   // 1/sqrt(64)

// ---- WMMA fragment helpers (layouts per CDNA5 ISA §7.12.2) -----
// A (16x32 bf16): lane m in half 0 holds K = {0..7, 16..23}; half 1 holds
// K = {8..15, 24..31}.  Two 16B chunks 32B apart.
static __device__ inline v16bf frag_a(const __bf16* rowBase, int half) {
  union { v4u u[2]; v16bf v; } f;
  f.u[0] = *(const v4u*)(rowBase + half * 8);
  f.u[1] = *(const v4u*)(rowBase + half * 8 + 16);
  return f.v;
}
// B (32x16 bf16): lane n holds column n, K = {half*8 + 0..7, 16 + half*8 + 0..7}.
static __device__ inline v16bf frag_b(const __bf16* rowBase, int half) {
  union { v4u u[2]; v16bf v; } f;
  f.u[0] = *(const v4u*)(rowBase + half * 8);
  f.u[1] = *(const v4u*)(rowBase + half * 8 + 16);
  return f.v;
}
static __device__ inline v8f wmma_bf16(v16bf a, v16bf b, v8f c) {
  // 8 args: (neg_a, A, neg_b, B, c_mod, C, reuse_a, reuse_b)
  return __builtin_amdgcn_wmma_f32_16x16x32_bf16(false, a, false, b,
                                                 (short)0, c, false, false);
}
static __device__ inline v8f vzero8() {
  v8f z = {0.f, 0.f, 0.f, 0.f, 0.f, 0.f, 0.f, 0.f};
  return z;
}

// ---- CDNA5 async global->LDS DMA (CDNA5 ISA) --------
// GV mode: LDS address comes from the VDST VGPR, global address from the 64-bit VADDR pair.
// The low 32 bits of a generic pointer to a __shared__ object are the LDS
// byte offset (aperture in the high bits), so we pass them as VDST.
static __device__ inline void async_ld16(const void* lds_ptr, const void* gptr) {
  unsigned lds_off = (unsigned)(unsigned long long)(uintptr_t)lds_ptr;
  unsigned long long ga = (unsigned long long)(uintptr_t)gptr;
  asm volatile("global_load_async_to_lds_b128 %0, %1, off"
               :: "v"(lds_off), "v"(ga) : "memory");
}
static __device__ inline void wait_async0() {
  asm volatile("s_wait_asynccnt 0x0" ::: "memory");
}

// ---------------------------------------------------------------------------
// Kernel 1: f32 -> bf16 conversion (x and weight matrices).
// ---------------------------------------------------------------------------
__global__ __launch_bounds__(256) void cvt_f32_bf16(const float* __restrict__ src,
                                                    __bf16* __restrict__ dst, int n) {
  int i = blockIdx.x * blockDim.x + threadIdx.x;
  if (i < n) dst[i] = (__bf16)src[i];
}

// ---------------------------------------------------------------------------
// Kernel 2/4: C[2048x512] = A[2048x512](bf16) * W[512x512]^T(bf16) + bias.
// 128x128 block per WG, 8 waves, each wave owns a 32x64 region (2x4 tiles).
// Double-buffered LDS (row stride padded to 40 bf16 => conflict-free
// ds_load_b128 fragment reads); next tile staged via async DMA while the
// current tile feeds the WMMAs; one barrier per K-step.
// ---------------------------------------------------------------------------
template <bool OUT_F32>
__global__ __launch_bounds__(256) void gemm_nt_bf16(const __bf16* __restrict__ A,
                                                    const __bf16* __restrict__ W,
                                                    const float* __restrict__ bias,
                                                    void* __restrict__ out) {
  __shared__ __bf16 As[2][128][40];  // 2 x 10 KB
  __shared__ __bf16 Ws[2][128][40];  // 2 x 10 KB

  const int tid  = threadIdx.x;
  const int wave = tid >> 5;
  const int lane = tid & 31;
  const int ln   = lane & 15;
  const int half = lane >> 4;
  const int m0   = blockIdx.x * 128;
  const int n0   = blockIdx.y * 128;
  const int wm   = (wave & 3) * 32;   // wave's M offset within block
  const int wn   = (wave >> 2) * 64;  // wave's N offset within block

  // Each thread stages 2 x 16B chunks of A and of W per K-tile.
  const int r0 = tid >> 2;                 // chunk 0 row (0..63 -> tid<256)
  const int c0 = (tid & 3) * 8;
  const int r1 = (tid + 256) >> 2;
  const int c1 = ((tid + 256) & 3) * 8;

  v8f acc[2][4];
#pragma unroll
  for (int mt = 0; mt < 2; ++mt)
#pragma unroll
    for (int nt = 0; nt < 4; ++nt) acc[mt][nt] = vzero8();

  // Prologue: async-stage K-tile 0 into buffer 0.
  async_ld16(&As[0][r0][c0], &A[(size_t)(m0 + r0) * DM + 0 + c0]);
  async_ld16(&As[0][r1][c1], &A[(size_t)(m0 + r1) * DM + 0 + c1]);
  async_ld16(&Ws[0][r0][c0], &W[(size_t)(n0 + r0) * DM + 0 + c0]);
  async_ld16(&Ws[0][r1][c1], &W[(size_t)(n0 + r1) * DM + 0 + c1]);
  wait_async0();
  __syncthreads();

  for (int kt = 0; kt < DM / 32; ++kt) {
    const int buf = kt & 1;
    // Kick off the async DMA for the next K-tile into the other buffer.
    if (kt + 1 < DM / 32) {
      const int k1 = (kt + 1) * 32;
      async_ld16(&As[buf ^ 1][r0][c0], &A[(size_t)(m0 + r0) * DM + k1 + c0]);
      async_ld16(&As[buf ^ 1][r1][c1], &A[(size_t)(m0 + r1) * DM + k1 + c1]);
      async_ld16(&Ws[buf ^ 1][r0][c0], &W[(size_t)(n0 + r0) * DM + k1 + c0]);
      async_ld16(&Ws[buf ^ 1][r1][c1], &W[(size_t)(n0 + r1) * DM + k1 + c1]);
    }
    // Warm L2/WGP$ for the tile after next (global_prefetch_b8).
    if (kt + 2 < DM / 32) {
      const int k2 = (kt + 2) * 32;
      __builtin_prefetch(&A[(size_t)(m0 + r0) * DM + k2 + c0], 0, 1);
      __builtin_prefetch(&W[(size_t)(n0 + r0) * DM + k2 + c0], 0, 1);
    }

    // Consume current buffer with WMMAs.
    v16bf af[2], bf[4];
#pragma unroll
    for (int mt = 0; mt < 2; ++mt) af[mt] = frag_a(&As[buf][wm + mt * 16 + ln][0], half);
#pragma unroll
    for (int nt = 0; nt < 4; ++nt) bf[nt] = frag_b(&Ws[buf][wn + nt * 16 + ln][0], half);
#pragma unroll
    for (int mt = 0; mt < 2; ++mt)
#pragma unroll
      for (int nt = 0; nt < 4; ++nt)
        acc[mt][nt] = wmma_bf16(af[mt], bf[nt], acc[mt][nt]);

    wait_async0();      // our next-tile DMA is done
    __syncthreads();    // everyone's DMA is done; reads of `buf` are done
  }

  // Store: C/D tile layout — VGPR i holds rows i (lanes 0-15) and i+8 (16-31).
#pragma unroll
  for (int mt = 0; mt < 2; ++mt) {
#pragma unroll
    for (int nt = 0; nt < 4; ++nt) {
      int n = n0 + wn + nt * 16 + ln;
      float bv = bias[n];
#pragma unroll
      for (int i = 0; i < 8; ++i) {
        int m = m0 + wm + mt * 16 + i + 8 * half;
        float v = acc[mt][nt][i] + bv;
        if (OUT_F32) ((float*)out)[(size_t)m * DM + n] = v;
        else         ((__bf16*)out)[(size_t)m * DM + n] = (__bf16)v;
      }
    }
  }
}

// ---------------------------------------------------------------------------
// Kernel 3: banded attention. One WG = (head, 64-query block).
// Keys needed: [q0-64, q0+127] -> 192 keys.  LDS ~136 KB (WGP has 320 KB).
//   Qs  [64][72]  bf16   Ks [192][72] bf16   Vt [64][200] bf16 (V transposed)
//   Ss  [64][200] f32    Ps [64][200] bf16
// ---------------------------------------------------------------------------
#define QS_LD 72
#define VT_LD 200
#define ATTN_SMEM_BYTES ((64*72 + 192*72 + 64*200) * 2 + 64*200*4 + 64*200*2)

__global__ __launch_bounds__(256) void attn_banded(const __bf16* __restrict__ Q,
                                                   const __bf16* __restrict__ K,
                                                   const __bf16* __restrict__ V,
                                                   __bf16* __restrict__ O) {
  extern __shared__ char smem[];
  __bf16* Qs = (__bf16*)smem;                       // 64 x 72
  __bf16* Ks = Qs + 64 * QS_LD;                     // 192 x 72
  __bf16* Vt = Ks + 192 * QS_LD;                    // 64 x 200 (Vt[d][key])
  float*  Ss = (float*)(Vt + 64 * VT_LD);           // 64 x 200
  __bf16* Ps = (__bf16*)(Ss + 64 * VT_LD);          // 64 x 200

  const int tid  = threadIdx.x;
  const int wave = tid >> 5;
  const int lane = tid & 31;
  const int ln   = lane & 15;
  const int half = lane >> 4;
  const int q0   = blockIdx.x * 64;
  const int hb   = blockIdx.y * 64;   // head's column base in [s,512]

  // ---- Phase 1: stage Q, K (row-major) and V (transposed) in LDS ----------
#pragma unroll
  for (int i = 0; i < 2; ++i) {               // Q: 64x64, 512 chunks
    int c = tid + i * 256;
    int r = c >> 3, cc = (c & 7) * 8;
    *(v4u*)&Qs[r * QS_LD + cc] = *(const v4u*)&Q[(size_t)(q0 + r) * DM + hb + cc];
  }
#pragma unroll
  for (int i = 0; i < 6; ++i) {               // K: 192x64, zero-fill OOB keys
    int c = tid + i * 256;
    int r = c >> 3, cc = (c & 7) * 8;
    int kpos = q0 - WIN + r;
    v4u val = {0u, 0u, 0u, 0u};
    if (kpos >= 0 && kpos < S_LEN)
      val = *(const v4u*)&K[(size_t)kpos * DM + hb + cc];
    *(v4u*)&Ks[r * QS_LD + cc] = val;
  }
  for (int i = 0; i < 48; ++i) {              // V: transpose into Vt[d][key]
    int c = i * 256 + tid;                    // consecutive tid -> consecutive d
    int j = c >> 6, d = c & 63;
    int kpos = q0 - WIN + j;
    __bf16 v = (__bf16)0.0f;
    if (kpos >= 0 && kpos < S_LEN) v = V[(size_t)kpos * DM + hb + d];
    Vt[d * VT_LD + j] = v;
  }
  __syncthreads();

  // ---- Phase 2: S[64x192] = Q * K^T, scale + band mask --------------------
  {
    const int mtile = wave >> 1;
    const int nbase = (wave & 1) * 6;
    v16bf a0 = frag_a(&Qs[(mtile * 16 + ln) * QS_LD + 0], half);
    v16bf a1 = frag_a(&Qs[(mtile * 16 + ln) * QS_LD + 32], half);
#pragma unroll
    for (int t = 0; t < 6; ++t) {
      int nt = nbase + t;
      v8f acc = vzero8();
      acc = wmma_bf16(a0, frag_b(&Ks[(nt * 16 + ln) * QS_LD + 0], half), acc);
      acc = wmma_bf16(a1, frag_b(&Ks[(nt * 16 + ln) * QS_LD + 32], half), acc);
      int colL = nt * 16 + ln;
      int colPos = q0 - WIN + colL;
#pragma unroll
      for (int i = 0; i < 8; ++i) {
        int rowL = mtile * 16 + i + 8 * half;
        int dd = colPos - (q0 + rowL);
        bool ok = (colPos >= 0) && (colPos < S_LEN) && (dd >= -WIN) && (dd <= WIN);
        Ss[rowL * VT_LD + colL] = ok ? acc[i] * ATT_SCALE : -1e30f;
      }
    }
  }
  __syncthreads();

  // ---- Phase 3: softmax over 192 cols; 4 lanes/row, wave shuffle reduce ---
  {
    int r = tid >> 2, part = tid & 3;
    int c0s = part * 48;
    float m = -1e30f;
    for (int c = c0s; c < c0s + 48; ++c) m = fmaxf(m, Ss[r * VT_LD + c]);
    m = fmaxf(m, __shfl_xor(m, 1, 32));
    m = fmaxf(m, __shfl_xor(m, 2, 32));
    float s = 0.f;
    for (int c = c0s; c < c0s + 48; ++c) {
      float e = __expf(Ss[r * VT_LD + c] - m);
      Ss[r * VT_LD + c] = e;
      s += e;
    }
    s += __shfl_xor(s, 1, 32);
    s += __shfl_xor(s, 2, 32);
    float inv = 1.0f / s;
    for (int c = c0s; c < c0s + 48; ++c)
      Ps[r * VT_LD + c] = (__bf16)(Ss[r * VT_LD + c] * inv);
  }
  __syncthreads();

  // ---- Phase 4: O[64x64] = P[64x192] * V[192x64] --------------------------
  {
    const int mtile = wave >> 1;
    const int nb    = (wave & 1) * 2;
    v8f oacc[2] = {vzero8(), vzero8()};
#pragma unroll
    for (int ks = 0; ks < 6; ++ks) {
      v16bf a = frag_a(&Ps[(mtile * 16 + ln) * VT_LD + ks * 32], half);
#pragma unroll
      for (int t = 0; t < 2; ++t) {
        // B[k=key][n=d] = V[key][d] = Vt[d][key]: contiguous per lane.
        v16bf b = frag_b(&Vt[((nb + t) * 16 + ln) * VT_LD + ks * 32], half);
        oacc[t] = wmma_bf16(a, b, oacc[t]);
      }
    }
#pragma unroll
    for (int t = 0; t < 2; ++t) {
      int col = hb + (nb + t) * 16 + ln;
#pragma unroll
      for (int i = 0; i < 8; ++i) {
        int row = q0 + mtile * 16 + i + 8 * half;
        O[(size_t)row * DM + col] = (__bf16)oacc[t][i];
      }
    }
  }
}

// ---------------------------------------------------------------------------
// Host launcher. ws layout (bytes): xb 2M | wq/wk/wv/wo 0.5M each | q/k/v 2M
// each | attn-out 2M  => 12 MB total.
// ---------------------------------------------------------------------------
extern "C" void kernel_launch(void* const* d_in, const int* in_sizes, int n_in,
                              void* d_out, int out_size, void* d_ws, size_t ws_size,
                              hipStream_t stream) {
  const float* x  = (const float*)d_in[0];
  const float* Wq = (const float*)d_in[1];
  const float* bq = (const float*)d_in[2];
  const float* Wk = (const float*)d_in[3];
  const float* bk = (const float*)d_in[4];
  const float* Wv = (const float*)d_in[5];
  const float* bv = (const float*)d_in[6];
  const float* Wo = (const float*)d_in[7];
  const float* bo = (const float*)d_in[8];

  char* p = (char*)d_ws;
  __bf16* xb  = (__bf16*)p; p += (size_t)S_LEN * DM * 2;
  __bf16* wqb = (__bf16*)p; p += (size_t)DM * DM * 2;
  __bf16* wkb = (__bf16*)p; p += (size_t)DM * DM * 2;
  __bf16* wvb = (__bf16*)p; p += (size_t)DM * DM * 2;
  __bf16* wob = (__bf16*)p; p += (size_t)DM * DM * 2;
  __bf16* qb  = (__bf16*)p; p += (size_t)S_LEN * DM * 2;
  __bf16* kb  = (__bf16*)p; p += (size_t)S_LEN * DM * 2;
  __bf16* vb  = (__bf16*)p; p += (size_t)S_LEN * DM * 2;
  __bf16* aob = (__bf16*)p; p += (size_t)S_LEN * DM * 2;

  cvt_f32_bf16<<<(S_LEN * DM) / 256, 256, 0, stream>>>(x, xb, S_LEN * DM);
  cvt_f32_bf16<<<(DM * DM) / 256, 256, 0, stream>>>(Wq, wqb, DM * DM);
  cvt_f32_bf16<<<(DM * DM) / 256, 256, 0, stream>>>(Wk, wkb, DM * DM);
  cvt_f32_bf16<<<(DM * DM) / 256, 256, 0, stream>>>(Wv, wvb, DM * DM);
  cvt_f32_bf16<<<(DM * DM) / 256, 256, 0, stream>>>(Wo, wob, DM * DM);

  dim3 g(S_LEN / 128, DM / 128);
  gemm_nt_bf16<false><<<g, 256, 0, stream>>>(xb, wqb, bq, qb);
  gemm_nt_bf16<false><<<g, 256, 0, stream>>>(xb, wkb, bk, kb);
  gemm_nt_bf16<false><<<g, 256, 0, stream>>>(xb, wvb, bv, vb);

  attn_banded<<<dim3(S_LEN / 64, 8), 256, ATTN_SMEM_BYTES, stream>>>(qb, kb, vb, aob);

  gemm_nt_bf16<true><<<g, 256, 0, stream>>>(aob, wob, bo, d_out);
}
